// MultiHeadGraphAttention_23862838297168
// MI455X (gfx1250) — matrix operand order repaired and checked by reading, not compiled, not run on hardware
//
#include <hip/hip_runtime.h>
#include <hip/hip_bf16.h>

// ---- problem constants (match reference) ----
#define N_NODES 4096
#define F_IN    256
#define F_OUT   64
#define N_HEAD  4

typedef __bf16 v16bf __attribute__((ext_vector_type(16)));
typedef __bf16 v8bf  __attribute__((ext_vector_type(8)));
typedef float  v8f   __attribute__((ext_vector_type(8)));

// ---------------------------------------------------------------------------
// Kernel 0a: convert node features h (f32 row-major) -> bf16 row-major.
// ---------------------------------------------------------------------------
__global__ void cvt_h_bf16(const float* __restrict__ h, __bf16* __restrict__ hb) {
    int idx = blockIdx.x * blockDim.x + threadIdx.x;   // < 4096*256
    hb[idx] = (__bf16)h[idx];
}

// ---------------------------------------------------------------------------
// Kernel 0b: pre-swizzle W into WMMA B-fragment lane layout (bf16 pairs).
// Layout: [head][kTile t:8][colTile n:4][lane:32][vgpr:8] as uint32
//   lane l owns column N = n*16 + (l&15), K base = t*32 + (l>>4)*16,
//   vgpr v holds K = base+2v (lo 16b) and base+2v+1 (hi 16b).
// ---------------------------------------------------------------------------
__global__ void pack_w_frags(const float* __restrict__ w, unsigned int* __restrict__ wb) {
    int idx = blockIdx.x * blockDim.x + threadIdx.x;   // < 4*8*4*32*8 = 32768
    int v  = idx & 7;
    int l  = (idx >> 3) & 31;
    int n  = (idx >> 8) & 3;
    int t  = (idx >> 10) & 7;
    int hd = idx >> 13;
    int k   = t * 32 + (l >> 4) * 16 + 2 * v;
    int col = n * 16 + (l & 15);
    float f0 = w[(hd * F_IN + k    ) * F_OUT + col];
    float f1 = w[(hd * F_IN + k + 1) * F_OUT + col];
    unsigned short u0 = __builtin_bit_cast(unsigned short, (__bf16)f0);
    unsigned short u1 = __builtin_bit_cast(unsigned short, (__bf16)f1);
    wb[idx] = ((unsigned int)u1 << 16) | (unsigned int)u0;
}

// ---------------------------------------------------------------------------
// Kernel 1: h_prime[h] = h @ W[h] via v_wmma_f32_16x16x32_bf16.
// grid = (32 row-blocks, 4 heads), 256 threads = 8 waves.
// Each wave: one 16-row strip x all 64 cols (4 C tiles), A-frag reuse x4.
// Packed W for the head (32 KB) staged in LDS.
// ---------------------------------------------------------------------------
__global__ void __launch_bounds__(256)
gat_hprime_wmma(const __bf16* __restrict__ hb,
                const unsigned int* __restrict__ wb,
                float* __restrict__ hp) {
    __shared__ __align__(16) unsigned int sB[8192];   // 32 KB packed B frags
    const int head = blockIdx.y;
    const int tid  = threadIdx.x;

    // stage packed W[head] into LDS: 2048 uint4, 8 per thread
    {
        const uint4* src = (const uint4*)(wb + head * 8192);
        uint4*       dst = (uint4*)sB;
#pragma unroll
        for (int r = 0; r < 8; ++r) dst[tid + r * 256] = src[tid + r * 256];
    }
    __syncthreads();

    const int wave    = tid >> 5;
    const int lane    = tid & 31;
    const int rowTile = blockIdx.x * 8 + wave;          // 0..255
    const int row     = rowTile * 16 + (lane & 15);     // A-frag row (M = lane%16)
    const int k0l     = (lane >> 4) * 16;               // A-frag K sub-base per half-wave

    v8f c0 = {}, c1 = {}, c2 = {}, c3 = {};

#pragma unroll
    for (int t = 0; t < 8; ++t) {                       // K = 256 in steps of 32
        // A fragment: two contiguous 16-byte bf16 loads per lane
        const __bf16* arow = hb + row * F_IN + t * 32 + k0l;
        v8bf lo = *(const v8bf*)(arow);
        v8bf hi = *(const v8bf*)(arow + 8);
        v16bf a;
#pragma unroll
        for (int e = 0; e < 8; ++e) { a[e] = lo[e]; a[8 + e] = hi[e]; }

        // B fragments: one 32-byte LDS read per tile per lane (pre-swizzled)
        const v16bf* bp = (const v16bf*)sB;             // 1 v16bf == lane's 8 uint32
        v16bf b0 = bp[(t * 4 + 0) * 32 + lane];
        v16bf b1 = bp[(t * 4 + 1) * 32 + lane];
        v16bf b2 = bp[(t * 4 + 2) * 32 + lane];
        v16bf b3 = bp[(t * 4 + 3) * 32 + lane];

        c0 = __builtin_amdgcn_wmma_f32_16x16x32_bf16(false, a, false, b0, (short)0, c0, false, false);
        c1 = __builtin_amdgcn_wmma_f32_16x16x32_bf16(false, a, false, b1, (short)0, c1, false, false);
        c2 = __builtin_amdgcn_wmma_f32_16x16x32_bf16(false, a, false, b2, (short)0, c2, false, false);
        c3 = __builtin_amdgcn_wmma_f32_16x16x32_bf16(false, a, false, b3, (short)0, c3, false, false);
    }

    // C/D layout: VGPR g -> M = g + 8*(lane>>4), N = lane&15
    const int rbase = rowTile * 16 + 8 * (lane >> 4);
    const int colb  = lane & 15;
#pragma unroll
    for (int g = 0; g < 8; ++g) {
        float* orow = hp + ((size_t)(head * N_NODES + rbase + g)) * F_OUT;
        orow[ 0 + colb] = c0[g];
        orow[16 + colb] = c1[g];
        orow[32 + colb] = c2[g];
        orow[48 + colb] = c3[g];
    }
}

// ---------------------------------------------------------------------------
// Kernel 2: src[h,n] = h_prime[h,n,:].a_src[h]; dst likewise. 16384 dots of 64.
// ---------------------------------------------------------------------------
__global__ void gat_logits(const float* __restrict__ hp,
                           const float* __restrict__ a_src,
                           const float* __restrict__ a_dst,
                           float* __restrict__ srcv, float* __restrict__ dstv) {
    int tid = blockIdx.x * blockDim.x + threadIdx.x;   // tid = head*4096 + node
    int hd  = tid >> 12;
    const float* row = hp + (size_t)tid * F_OUT;
    float s = 0.f, d = 0.f;
#pragma unroll
    for (int f = 0; f < F_OUT; ++f) {
        float p = row[f];
        s = fmaf(p, a_src[hd * F_OUT + f], s);
        d = fmaf(p, a_dst[hd * F_OUT + f], d);
    }
    srcv[tid] = s;
    dstv[tid] = d;
}

// ---------------------------------------------------------------------------
// Kernel 3: adjacency-driven online-softmax aggregation (all 4 heads fused).
// One wave per node; lanes stream 32 adjacency bytes/iter, ballot valid
// neighbors, flash-style rescale+accumulate. Lane owns features {l, l+32}.
// adj read exactly once (16.7 MB); h_prime gathers hit L2 (4 MB resident).
// ---------------------------------------------------------------------------
__global__ void __launch_bounds__(256)
gat_aggregate(const unsigned char* __restrict__ adj,
              const float* __restrict__ hp,
              const float* __restrict__ srcv, const float* __restrict__ dstv,
              const float* __restrict__ bias, float* __restrict__ out) {
    const int wave = threadIdx.x >> 5;
    const int lane = threadIdx.x & 31;
    const int i    = blockIdx.x * 8 + wave;            // node id
    const unsigned char* arow = adj + (size_t)i * N_NODES;

    float si[N_HEAD];
#pragma unroll
    for (int hd = 0; hd < N_HEAD; ++hd) si[hd] = srcv[hd * N_NODES + i];

    float m[N_HEAD], ss[N_HEAD], acc0[N_HEAD], acc1[N_HEAD];
#pragma unroll
    for (int hd = 0; hd < N_HEAD; ++hd) {
        m[hd] = -__builtin_inff(); ss[hd] = 0.f; acc0[hd] = 0.f; acc1[hd] = 0.f;
    }

    for (int j0 = 0; j0 < N_NODES; j0 += 32) {
        __builtin_prefetch(arow + j0 + 256, 0, 1);     // global_prefetch_b8 ahead
        unsigned char av = arow[j0 + lane];
#if __has_builtin(__builtin_amdgcn_ballot_w32)
        unsigned int mask = __builtin_amdgcn_ballot_w32(av != 0);
#else
        unsigned int mask = (unsigned int)__ballot(av != 0);
#endif
        while (mask) {
            int b = __builtin_ctz(mask);
            mask &= mask - 1;
            int j = j0 + b;
#pragma unroll
            for (int hd = 0; hd < N_HEAD; ++hd) {
                float sc = si[hd] + dstv[hd * N_NODES + j];
                sc = sc >= 0.f ? sc : 0.2f * sc;        // leaky_relu(0.2)
                float mn    = fmaxf(m[hd], sc);
                float scale = __expf(m[hd] - mn);       // exp(-inf)=0 on first hit
                float e     = __expf(sc - mn);
                const float* prow = hp + ((size_t)(hd * N_NODES + j)) * F_OUT;
                float p0 = prow[lane];
                float p1 = prow[lane + 32];
                ss[hd]   = ss[hd]   * scale + e;
                acc0[hd] = acc0[hd] * scale + e * p0;
                acc1[hd] = acc1[hd] * scale + e * p1;
                m[hd] = mn;
            }
        }
    }

    float b0 = bias[lane], b1 = bias[lane + 32];
#pragma unroll
    for (int hd = 0; hd < N_HEAD; ++hd) {
        float inv = 1.f / ss[hd];                       // self-loop => ss > 0
        float* orow = out + ((size_t)i * N_HEAD + hd) * F_OUT;  // [n, H, f_out]
        orow[lane]      = acc0[hd] * inv + b0;
        orow[lane + 32] = acc1[hd] * inv + b1;
    }
}

// ---------------------------------------------------------------------------
// launch
// ---------------------------------------------------------------------------
extern "C" void kernel_launch(void* const* d_in, const int* in_sizes, int n_in,
                              void* d_out, int out_size, void* d_ws, size_t ws_size,
                              hipStream_t stream) {
    (void)in_sizes; (void)n_in; (void)out_size; (void)ws_size;
    const float*         h     = (const float*)d_in[0];
    const unsigned char* adj   = (const unsigned char*)d_in[1];  // jnp bool = 1 byte
    const float*         w     = (const float*)d_in[2];
    const float*         a_src = (const float*)d_in[3];
    const float*         a_dst = (const float*)d_in[4];
    const float*         bias  = (const float*)d_in[5];
    float*               out   = (float*)d_out;

    // workspace carve-up (~6.3 MB)
    float*        hp   = (float*)d_ws;                       // 4*4096*64 f32 (4 MB)
    float*        srcv = hp + (size_t)N_HEAD * N_NODES * F_OUT;
    float*        dstv = srcv + (size_t)N_HEAD * N_NODES;
    __bf16*       hb   = (__bf16*)(dstv + (size_t)N_HEAD * N_NODES);   // 2 MB
    unsigned int* wb   = (unsigned int*)(hb + (size_t)N_NODES * F_IN); // 128 KB

    cvt_h_bf16  <<<(N_NODES * F_IN) / 256, 256, 0, stream>>>(h, hb);
    pack_w_frags<<<(N_HEAD * F_IN * F_OUT / 2) / 256, 256, 0, stream>>>(w, wb);

    dim3 g1(N_NODES / 128, N_HEAD);                          // 8 waves x 16 rows = 128 rows/blk
    gat_hprime_wmma<<<g1, 256, 0, stream>>>(hb, wb, hp);

    gat_logits  <<<(N_HEAD * N_NODES) / 256, 256, 0, stream>>>(hp, a_src, a_dst, srcv, dstv);
    gat_aggregate<<<N_NODES / 8, 256, 0, stream>>>(adj, hp, srcv, dstv, bias, out);
}
